// PatchResNet_24567212933599
// MI455X (gfx1250) — matrix operand order repaired and checked by reading, not compiled, not run on hardware
//
#include <hip/hip_runtime.h>
#include <hip/hip_bf16.h>

typedef __attribute__((ext_vector_type(16))) _Float16 v16h;
typedef __attribute__((ext_vector_type(8)))  _Float16 v8h;
typedef __attribute__((ext_vector_type(8)))  float    v8f;

// ---------------------------------------------------------------------------
// BN fold: scale = gamma * rsqrt(var+eps); bias = beta - mean*scale
// ---------------------------------------------------------------------------
__global__ void bn_fold_kernel(const float* __restrict__ gamma, const float* __restrict__ beta,
                               const float* __restrict__ mean, const float* __restrict__ var,
                               float* __restrict__ scale, float* __restrict__ bias, int C) {
    int i = blockIdx.x * blockDim.x + threadIdx.x;
    if (i >= C) return;
    float s = gamma[i] * rsqrtf(var[i] + 1e-5f);
    scale[i] = s;
    bias[i]  = beta[i] - mean[i] * s;
}

// ---------------------------------------------------------------------------
// Weight repack: OIHW f32 -> [O][kh][kw][I] f16 (K-dim = (kh,kw,ci), ci innermost)
// ---------------------------------------------------------------------------
__global__ void repack_w_kernel(const float* __restrict__ w, _Float16* __restrict__ out,
                                int O, int I, int KHW) {
    int idx = blockIdx.x * blockDim.x + threadIdx.x;
    int total = O * I * KHW;
    if (idx >= total) return;
    int o = idx / (KHW * I);
    int r = idx - o * (KHW * I);
    int t = r / I;
    int i = r - t * I;
    out[idx] = (_Float16)w[(o * I + i) * KHW + t];
}

// ---------------------------------------------------------------------------
// Fused: patch extraction + conv1(7x7, Cin=1, pad3) + BN + ReLU + maxpool(3x3,s2,p1)
// out: [144 patches][25][25][64] f16, channels innermost
// ---------------------------------------------------------------------------
__global__ __launch_bounds__(256) void conv1_pool_kernel(
    const float* __restrict__ x,      // (12,12)
    const float* __restrict__ w,      // (64,1,7,7)
    const float* __restrict__ scale, const float* __restrict__ bias,
    _Float16* __restrict__ out) {
    __shared__ float xs[144];
    __shared__ float ws[64 * 49];
    int tid = threadIdx.x;
    if (tid < 144) xs[tid] = x[tid];
    for (int i = tid; i < 64 * 49; i += 256) ws[i] = w[i];
    __syncthreads();

    int gid = blockIdx.x * 256 + tid;     // exact grid: 144*25*25*64 threads
    int o = gid & 63;
    int t = gid >> 6;
    int px = t % 25; t /= 25;
    int py = t % 25; t /= 25;
    int p  = t;                            // patch 0..143
    int pi = p / 12, pj = p % 12;

    float best = 0.0f;                     // relu() >= 0 and window never empty
    for (int dy = -1; dy <= 1; ++dy) {
        int wy = 2 * py + dy; if (wy < 0 || wy >= 50) continue;
        for (int dx = -1; dx <= 1; ++dx) {
            int wx = 2 * px + dx; if (wx < 0 || wx >= 50) continue;
            float acc = 0.0f;
            for (int kh = 0; kh < 7; ++kh) {
                int ry = pi + wy + kh - 28; if (ry < 0 || ry >= 12) continue;
                for (int kw = 0; kw < 7; ++kw) {
                    int rx = pj + wx + kw - 28; if (rx < 0 || rx >= 12) continue;
                    acc += ws[o * 49 + kh * 7 + kw] * xs[ry * 12 + rx];
                }
            }
            float v = acc * scale[o] + bias[o];
            v = v > 0.f ? v : 0.f;
            best = best > v ? best : v;
        }
    }
    out[gid] = (_Float16)best;
}

// ---------------------------------------------------------------------------
// Implicit-GEMM conv via v_wmma_f32_16x16x32_f16 + fused BN / residual / ReLU.
// Fully compile-time geometry. One wave computes a 32(Cout) x 32(pixel) tile:
// 2 A-fragments x 2 B-fragments -> 4 WMMAs per 32-wide K step.
// in : [144*HIN*HIN][CIN] f16   wt: [COUT][KH*KH*CIN] f16
// out: [Npix][COUT] f16 ; Npix = 144*HOUT*HOUT (multiple of 16)
// ---------------------------------------------------------------------------
template<int KH, int STRIDE, int PAD, int CIN, int COUT, int HIN, int HOUT,
         bool RELU, bool HAS_RES>
__global__ __launch_bounds__(128) void conv_wmma(
    const _Float16* __restrict__ in, const _Float16* __restrict__ wt,
    const float* __restrict__ scale, const float* __restrict__ bias,
    const _Float16* __restrict__ res, _Float16* __restrict__ out) {
    constexpr int Npix = 144 * HOUT * HOUT;
    constexpr int Ktot = KH * KH * CIN;

    const int lane = threadIdx.x;                                 // 0..31
    const int t2   = blockIdx.x * blockDim.y + threadIdx.y;       // 32-pixel tile
    const int n0   = t2 * 32;
    if (n0 >= Npix) return;                                       // wave-uniform
    const bool hasB = (n0 + 16) < Npix;                           // 2nd pixel tile valid
    const int cout0 = blockIdx.y * 32;
    const int l15 = lane & 15;
    const int hi  = lane >> 4;
    const int kb  = hi * 8;                                       // B-fragment K offset (elems 0-7 -> K=hi*8.., 8-15 -> +16)

    // decode both pixel columns (compile-time divisors -> mul/shift)
    int nA = n0 + l15;
    int nB = hasB ? (nA + 16) : nA;                               // clamp keeps loads safe
    int imgA = nA / (HOUT * HOUT), rA = nA - imgA * (HOUT * HOUT);
    int imgB = nB / (HOUT * HOUT), rB = nB - imgB * (HOUT * HOUT);
    int oyA = rA / HOUT, oxA = rA - oyA * HOUT;
    int oyB = rB / HOUT, oxB = rB - oyB * HOUT;
    const int iy0A = oyA * STRIDE - PAD, ix0A = oxA * STRIDE - PAD;
    const int iy0B = oyB * STRIDE - PAD, ix0B = oxB * STRIDE - PAD;

    // A-fragment row pointers (elems 0-7 -> K=k0+hi*8.., elems 8-15 -> +16)
    const _Float16* aptr0 = wt + (size_t)(cout0 + l15) * Ktot + hi * 8;
    const _Float16* aptr1 = aptr0 + (size_t)16 * Ktot;

    v8f acc00 = {}, acc01 = {}, acc10 = {}, acc11 = {};           // [coutTile][pixTile]

#pragma unroll 1
    for (int t = 0; t < KH * KH; ++t) {                           // conv tap
        const int kh = t / KH, kw = t - kh * KH;
        const int iyA = iy0A + kh, ixA = ix0A + kw;
        const int iyB = iy0B + kh, ixB = ix0B + kw;
        const bool inA = (iyA >= 0 && iyA < HIN && ixA >= 0 && ixA < HIN);
        const bool inB = (iyB >= 0 && iyB < HIN && ixB >= 0 && ixB < HIN);
        const _Float16* bpA = in + (((size_t)imgA * HIN + iyA) * HIN + ixA) * CIN + kb;
        const _Float16* bpB = in + (((size_t)imgB * HIN + iyB) * HIN + ixB) * CIN + kb;
        const int kt = t * CIN;
        for (int c0 = 0; c0 < CIN; c0 += 32) {                    // K step of 32
            v16h a0, a1, b0 = {}, b1 = {};
            ((v8h*)&a0)[0] = *(const v8h*)(aptr0 + kt + c0);
            ((v8h*)&a0)[1] = *(const v8h*)(aptr0 + kt + c0 + 16);
            ((v8h*)&a1)[0] = *(const v8h*)(aptr1 + kt + c0);
            ((v8h*)&a1)[1] = *(const v8h*)(aptr1 + kt + c0 + 16);
            if (inA) { ((v8h*)&b0)[0] = *(const v8h*)(bpA + c0);
                       ((v8h*)&b0)[1] = *(const v8h*)(bpA + c0 + 16); }
            if (inB) { ((v8h*)&b1)[0] = *(const v8h*)(bpB + c0);
                       ((v8h*)&b1)[1] = *(const v8h*)(bpB + c0 + 16); }
            acc00 = __builtin_amdgcn_wmma_f32_16x16x32_f16(false, a0, false, b0, (short)0, acc00, false, false);
            acc01 = __builtin_amdgcn_wmma_f32_16x16x32_f16(false, a0, false, b1, (short)0, acc01, false, false);
            acc10 = __builtin_amdgcn_wmma_f32_16x16x32_f16(false, a1, false, b0, (short)0, acc10, false, false);
            acc11 = __builtin_amdgcn_wmma_f32_16x16x32_f16(false, a1, false, b1, (short)0, acc11, false, false);
        }
    }

    // epilogue: BN scale/bias (+residual) (+relu); 16B packed stores
    const int mo = hi * 8;                                        // C/D: m = r + 8*hi
#pragma unroll
    for (int mt = 0; mt < 2; ++mt) {
        const v8f& cA = mt ? acc10 : acc00;
        const v8f& cB = mt ? acc11 : acc01;
        const int cb = cout0 + mt * 16 + mo;
        size_t oA = (size_t)nA * COUT + cb;
        size_t oB = (size_t)nB * COUT + cb;
        v8h pA, pB;
#pragma unroll
        for (int r = 0; r < 8; ++r) {
            float s = scale[cb + r], bi = bias[cb + r];
            float vA = cA[r] * s + bi;
            float vB = cB[r] * s + bi;
            if constexpr (HAS_RES) { vA += (float)res[oA + r]; vB += (float)res[oB + r]; }
            if constexpr (RELU)    { vA = vA > 0.f ? vA : 0.f; vB = vB > 0.f ? vB : 0.f; }
            pA[r] = (_Float16)vA; pB[r] = (_Float16)vB;
        }
        *(v8h*)(out + oA) = pA;
        if (hasB) *(v8h*)(out + oB) = pB;
    }
}

// ---------------------------------------------------------------------------
// adaptive avgpool (4x4) + FC(512->2); out[cls*144 + patch] f32
// ---------------------------------------------------------------------------
__global__ void avgpool_fc_kernel(const _Float16* __restrict__ act,  // [144][16][512]
                                  const float* __restrict__ fc_w,    // [2][512]
                                  const float* __restrict__ fc_b,
                                  float* __restrict__ out) {
    int p = blockIdx.x, lane = threadIdx.x;
    const _Float16* a = act + (size_t)p * 16 * 512;
    float d0 = 0.f, d1 = 0.f;
    for (int c = lane; c < 512; c += 32) {
        float s = 0.f;
        for (int sp = 0; sp < 16; ++sp) s += (float)a[sp * 512 + c];
        s *= (1.0f / 16.0f);
        d0 += s * fc_w[c];
        d1 += s * fc_w[512 + c];
    }
    for (int off = 16; off > 0; off >>= 1) {
        d0 += __shfl_down(d0, off, 32);
        d1 += __shfl_down(d1, off, 32);
    }
    if (lane == 0) { out[p] = d0 + fc_b[0]; out[144 + p] = d1 + fc_b[1]; }
}

// ---------------------------------------------------------------------------
// Host side
// ---------------------------------------------------------------------------
static const int O_[20] = {64, 64,64, 64,64, 128,128,128, 128,128,
                           256,256,256, 256,256, 512,512,512, 512,512};
static const int I_[20] = {1,  64,64, 64,64, 64,128,64, 128,128,
                           128,256,128, 256,256, 256,512,256, 512,512};
static const int K_[20] = {7,  3,3, 3,3, 3,3,1, 3,3, 3,3,1, 3,3, 3,3,1, 3,3};

static const size_t SC_OFF  = 0;
static const size_t BI_OFF  = 19456;
static const size_t WPK_OFF = 38912;
static const size_t P0_OFF  = 22354176;
static const size_t ACT_B   = 11520000;   // 144*25*25*64 halves * 2B

template<int KH, int ST, int PD, int CI, int CO, int HI, int HO, bool RL, bool RS>
static inline void launch_conv(const _Float16* in, const _Float16* wt, const float* sc,
                               const float* bi, const _Float16* res, _Float16* out,
                               hipStream_t stream) {
    constexpr int Np = 144 * HO * HO;
    constexpr int tiles2 = (Np + 31) / 32;
    dim3 grid((tiles2 + 3) / 4, CO / 32), block(32, 4);
    conv_wmma<KH, ST, PD, CI, CO, HI, HO, RL, RS>
        <<<grid, block, 0, stream>>>(in, wt, sc, bi, res, out);
}

extern "C" void kernel_launch(void* const* d_in, const int* in_sizes, int n_in,
                              void* d_out, int out_size, void* d_ws, size_t ws_size,
                              hipStream_t stream) {
    (void)out_size; (void)ws_size;
    const float* conv_w[20]; const float *bn_g[20], *bn_b[20], *bn_m[20], *bn_v[20];
    const float *fc_w = nullptr, *fc_b = nullptr, *x = nullptr;
    auto F = [&](int i) { return (const float*)d_in[i]; };

    bool sorted_order = !(n_in > 0 && in_sizes[0] == 144);
    if (sorted_order) {
        // jax tree-flatten (keys sorted): params{bn1{beta,gamma,mean,var},conv1,fc_b,fc_w,
        // layer*{bn1,bn2,conv1,conv2[,down_bn,down_conv]}}, x
        bn_b[0]=F(0); bn_g[0]=F(1); bn_m[0]=F(2); bn_v[0]=F(3);
        conv_w[0]=F(4); fc_b=F(5); fc_w=F(6);
        int idx = 7, ci = 1;
        for (int l = 0; l < 4; ++l) for (int b = 0; b < 2; ++b) {
            bool dn = (l > 0 && b == 0);
            int c1 = ci, c2 = ci + 1;
            bn_b[c1]=F(idx+0); bn_g[c1]=F(idx+1); bn_m[c1]=F(idx+2); bn_v[c1]=F(idx+3);
            bn_b[c2]=F(idx+4); bn_g[c2]=F(idx+5); bn_m[c2]=F(idx+6); bn_v[c2]=F(idx+7);
            conv_w[c1]=F(idx+8); conv_w[c2]=F(idx+9); idx += 10;
            if (dn) { int cd=ci+2;
                bn_b[cd]=F(idx+0); bn_g[cd]=F(idx+1); bn_m[cd]=F(idx+2); bn_v[cd]=F(idx+3);
                conv_w[cd]=F(idx+4); idx += 5; }
            ci += dn ? 3 : 2;
        }
        x = F(idx);
    } else {
        // insertion-order flatten: x, conv1, bn1{gamma,beta,mean,var}, layers..., fc_w, fc_b
        x = F(0); conv_w[0]=F(1);
        bn_g[0]=F(2); bn_b[0]=F(3); bn_m[0]=F(4); bn_v[0]=F(5);
        int idx = 6, ci = 1;
        for (int l = 0; l < 4; ++l) for (int b = 0; b < 2; ++b) {
            bool dn = (l > 0 && b == 0);
            int c1 = ci, c2 = ci + 1;
            conv_w[c1]=F(idx); bn_g[c1]=F(idx+1); bn_b[c1]=F(idx+2); bn_m[c1]=F(idx+3); bn_v[c1]=F(idx+4);
            conv_w[c2]=F(idx+5); bn_g[c2]=F(idx+6); bn_b[c2]=F(idx+7); bn_m[c2]=F(idx+8); bn_v[c2]=F(idx+9);
            idx += 10;
            if (dn) { int cd=ci+2;
                conv_w[cd]=F(idx); bn_g[cd]=F(idx+1); bn_b[cd]=F(idx+2); bn_m[cd]=F(idx+3); bn_v[cd]=F(idx+4);
                idx += 5; }
            ci += dn ? 3 : 2;
        }
        fc_w = F(idx); fc_b = F(idx + 1);
    }

    char* ws = (char*)d_ws;
    float*     sc_all = (float*)(ws + SC_OFF);
    float*     bi_all = (float*)(ws + BI_OFF);
    _Float16*  wpk    = (_Float16*)(ws + WPK_OFF);
    _Float16*  P0     = (_Float16*)(ws + P0_OFF);
    _Float16*  P1     = (_Float16*)(ws + P0_OFF + ACT_B);
    _Float16*  T      = (_Float16*)(ws + P0_OFF + 2 * ACT_B);
    _Float16*  S      = (_Float16*)(ws + P0_OFF + 3 * ACT_B);

    int chOff[20]; size_t wOff[20]; { int a = 0; size_t wa = 0;
        for (int c = 0; c < 20; ++c) { chOff[c] = a; a += O_[c]; }
        for (int c = 1; c < 20; ++c) { wOff[c] = wa; wa += (size_t)O_[c]*I_[c]*K_[c]*K_[c]; } }

    for (int c = 0; c < 20; ++c)
        bn_fold_kernel<<<(O_[c] + 63) / 64, 64, 0, stream>>>(
            bn_g[c], bn_b[c], bn_m[c], bn_v[c], sc_all + chOff[c], bi_all + chOff[c], O_[c]);
    for (int c = 1; c < 20; ++c) {
        int tot = O_[c] * I_[c] * K_[c] * K_[c];
        repack_w_kernel<<<(tot + 255) / 256, 256, 0, stream>>>(
            conv_w[c], wpk + wOff[c], O_[c], I_[c], K_[c] * K_[c]);
    }

    conv1_pool_kernel<<<144 * 25 * 25 * 64 / 256, 256, 0, stream>>>(
        x, conv_w[0], sc_all + chOff[0], bi_all + chOff[0], P0);

    auto W = [&](int c) { return wpk + wOff[c]; };
    auto SC = [&](int c) { return sc_all + chOff[c]; };
    auto BI = [&](int c) { return bi_all + chOff[c]; };

    // layer1 (64ch, 25x25)
    launch_conv<3,1,1, 64, 64,25,25,true ,false>(P0, W(1),  SC(1),  BI(1),  nullptr, T,  stream);
    launch_conv<3,1,1, 64, 64,25,25,true ,true >(T,  W(2),  SC(2),  BI(2),  P0,      P1, stream);
    launch_conv<3,1,1, 64, 64,25,25,true ,false>(P1, W(3),  SC(3),  BI(3),  nullptr, T,  stream);
    launch_conv<3,1,1, 64, 64,25,25,true ,true >(T,  W(4),  SC(4),  BI(4),  P1,      P0, stream);
    // layer2 (128ch, 25->13)
    launch_conv<1,2,0, 64,128,25,13,false,false>(P0, W(7),  SC(7),  BI(7),  nullptr, S,  stream);
    launch_conv<3,2,1, 64,128,25,13,true ,false>(P0, W(5),  SC(5),  BI(5),  nullptr, T,  stream);
    launch_conv<3,1,1,128,128,13,13,true ,true >(T,  W(6),  SC(6),  BI(6),  S,       P1, stream);
    launch_conv<3,1,1,128,128,13,13,true ,false>(P1, W(8),  SC(8),  BI(8),  nullptr, T,  stream);
    launch_conv<3,1,1,128,128,13,13,true ,true >(T,  W(9),  SC(9),  BI(9),  P1,      P0, stream);
    // layer3 (256ch, 13->7)
    launch_conv<1,2,0,128,256,13, 7,false,false>(P0, W(12), SC(12), BI(12), nullptr, S,  stream);
    launch_conv<3,2,1,128,256,13, 7,true ,false>(P0, W(10), SC(10), BI(10), nullptr, T,  stream);
    launch_conv<3,1,1,256,256, 7, 7,true ,true >(T,  W(11), SC(11), BI(11), S,       P1, stream);
    launch_conv<3,1,1,256,256, 7, 7,true ,false>(P1, W(13), SC(13), BI(13), nullptr, T,  stream);
    launch_conv<3,1,1,256,256, 7, 7,true ,true >(T,  W(14), SC(14), BI(14), P1,      P0, stream);
    // layer4 (512ch, 7->4)
    launch_conv<1,2,0,256,512, 7, 4,false,false>(P0, W(17), SC(17), BI(17), nullptr, S,  stream);
    launch_conv<3,2,1,256,512, 7, 4,true ,false>(P0, W(15), SC(15), BI(15), nullptr, T,  stream);
    launch_conv<3,1,1,512,512, 4, 4,true ,true >(T,  W(16), SC(16), BI(16), S,       P1, stream);
    launch_conv<3,1,1,512,512, 4, 4,true ,false>(P1, W(18), SC(18), BI(18), nullptr, T,  stream);
    launch_conv<3,1,1,512,512, 4, 4,true ,true >(T,  W(19), SC(19), BI(19), P1,      P0, stream);

    avgpool_fc_kernel<<<144, 32, 0, stream>>>(P0, fc_w, fc_b, (float*)d_out);
}
